// UnifiedRingStarBlock_46179488367248
// MI455X (gfx1250) — matrix operand order repaired and checked by reading, not compiled, not run on hardware
//
#include <hip/hip_runtime.h>
#include <hip/hip_bf16.h>
#include <math.h>

// ---------------- problem constants ----------------
#define BB 32
#define LL 720
#define NN 512
#define HH 16
#define DD 64
#define TOPK 8
#define ROWS (BB*LL)          // 23040
#define TILES (ROWS/16)       // 1440

typedef __attribute__((ext_vector_type(8)))  float  v8f;
typedef __attribute__((ext_vector_type(8)))  __bf16 v8bf;
typedef __attribute__((ext_vector_type(16))) __bf16 v16bf;

// ---------------- helpers ----------------
__device__ inline void split_bf16(float f, unsigned short& hi, unsigned short& lo) {
    __bf16 h = (__bf16)f;
    float  fh = (float)h;
    __bf16 l = (__bf16)(f - fh);
    hi = __builtin_bit_cast(unsigned short, h);
    lo = __builtin_bit_cast(unsigned short, l);
}

// Swizzle within a 32-wide K block so that a lane reading 16 contiguous bf16
// gets exactly the ISA 16-bit A-matrix register layout:
//   lane group g needs K order: [8g..8g+7, 16+8g..16+8g+7]
// position of K value kl in the 32-element block: swap the middle two 8-blocks.
__device__ inline int swz32(int kl) {
    int blk = kl >> 3;
    return (blk == 1) ? (kl + 8) : ((blk == 2) ? (kl - 8) : kl);
}

__device__ inline v16bf load_v16(const unsigned short* p) {
    v8bf a = *reinterpret_cast<const v8bf*>(p);
    v8bf b = *reinterpret_cast<const v8bf*>(p + 8);
    return __builtin_shufflevector(a, b, 0,1,2,3,4,5,6,7,8,9,10,11,12,13,14,15);
}

__device__ inline v8f wmma_bf16(v16bf a, v16bf b, v8f c) {
    return __builtin_amdgcn_wmma_f32_16x16x32_bf16(false, a, false, b, (short)0, c, false, false);
}

__device__ inline float wave_sum(float v) {
    #pragma unroll
    for (int o = 16; o > 0; o >>= 1) v += __shfl_xor(v, o, 32);
    return v;
}

__device__ inline float gelu_exact(float a) {
    return 0.5f * a * (1.0f + erff(a * 0.70710678118654752440f));
}

// ---------------- k0: transpose + bf16 split of Wg (1024x512) and Wf (512x512) ----------------
__global__ void k0_prep(const float* __restrict__ Wg, const float* __restrict__ Wf,
                        unsigned short* __restrict__ wgT_hi, unsigned short* __restrict__ wgT_lo,
                        unsigned short* __restrict__ wfT_hi, unsigned short* __restrict__ wfT_lo) {
    int i = blockIdx.x * 256 + threadIdx.x;
    if (i < 512 * 1024) {                       // WgT[n][k] = Wg[k][n]
        int n = i >> 10, k = i & 1023;
        unsigned short hi, lo; split_bf16(Wg[k * 512 + n], hi, lo);
        wgT_hi[i] = hi; wgT_lo[i] = lo;
    } else {
        int j = i - 512 * 1024;
        if (j < 512 * 512) {                    // WfT[n][k] = Wf[k][n]
            int n = j >> 9, k = j & 511;
            unsigned short hi, lo; split_bf16(Wf[k * 512 + n], hi, lo);
            wfT_hi[j] = hi; wfT_lo[j] = lo;
        }
    }
}

// ---------------- k1: Q = ve@Wq+bq, K = ve@Wk+bk  (512x16) ----------------
__global__ void k1_qk(const float* __restrict__ ve,
                      const float* __restrict__ Wq, const float* __restrict__ bq,
                      const float* __restrict__ Wk, const float* __restrict__ bk,
                      float* __restrict__ Qg, float* __restrict__ Kg) {
    int n = blockIdx.x * 256 + threadIdx.x;
    if (n >= NN) return;
    float e[HH];
    #pragma unroll
    for (int j = 0; j < HH; ++j) e[j] = ve[n * HH + j];
    for (int h = 0; h < HH; ++h) {
        float q = bq[h], k = bk[h];
        #pragma unroll
        for (int j = 0; j < HH; ++j) { q += e[j] * Wq[j * HH + h]; k += e[j] * Wk[j * HH + h]; }
        Qg[n * HH + h] = q; Kg[n * HH + h] = k;
    }
}

// ---------------- k2: per-row top-8 of Q@K^T (diag masked) + softmax ----------------
__global__ void k2_topk(const float* __restrict__ Qg, const float* __restrict__ Kg,
                        float* __restrict__ w8, int* __restrict__ idx8) {
    int n = blockIdx.x * 256 + threadIdx.x;
    if (n >= NN) return;
    float q[HH];
    #pragma unroll
    for (int j = 0; j < HH; ++j) q[j] = Qg[n * HH + j];
    float tv[TOPK]; int ti[TOPK];
    #pragma unroll
    for (int k = 0; k < TOPK; ++k) { tv[k] = -3.0e38f; ti[k] = 0; }
    for (int col = 0; col < NN; ++col) {
        if (col == n) continue;                 // diagonal masked to -1e9: never in top-8
        float d = 0.f;
        #pragma unroll
        for (int j = 0; j < HH; ++j) d += q[j] * Kg[col * HH + j];
        if (d > tv[TOPK - 1]) {                 // strict > : stable ties (lower index first)
            int p = TOPK - 1;
            while (p > 0 && d > tv[p - 1]) { tv[p] = tv[p - 1]; ti[p] = ti[p - 1]; --p; }
            tv[p] = d; ti[p] = col;
        }
    }
    float mx = tv[0], sum = 0.f, wv[TOPK];
    #pragma unroll
    for (int k = 0; k < TOPK; ++k) { wv[k] = expf(tv[k] - mx); sum += wv[k]; }
    float inv = 1.f / sum;
    #pragma unroll
    for (int k = 0; k < TOPK; ++k) { w8[n * TOPK + k] = wv[k] * inv; idx8[n * TOPK + k] = ti[k]; }
}

// ---------------- k3: s[b,l] = x[b,l,:]@Wscore + bscore  (one wave per row) ----------------
__global__ void k3_score(const float* __restrict__ x, const float* __restrict__ Wscore,
                         const float* __restrict__ bscore, float* __restrict__ s) {
    int gid = blockIdx.x * 256 + threadIdx.x;
    int row = gid >> 5, lane = gid & 31;
    if (row >= ROWS) return;
    const float* xr = x + (size_t)row * NN;
    float acc = 0.f;
    #pragma unroll
    for (int j = 0; j < 16; ++j) { int nn = j * 32 + lane; acc += xr[nn] * Wscore[nn]; }
    acc = wave_sum(acc);
    if (lane == 0) s[row] = acc + bscore[0];
}

// ---------------- k4: softmax over L, pooled center, 3-layer MLP -> center[b][512] ----------------
__global__ void k4_center(const float* __restrict__ x, const float* __restrict__ s,
                          const float* __restrict__ Wc1, const float* __restrict__ bc1,
                          const float* __restrict__ Wc2, const float* __restrict__ bc2,
                          const float* __restrict__ Wcn, const float* __restrict__ bcn,
                          float* __restrict__ center) {
    __shared__ float sh_s[LL];
    __shared__ float sh_red[256];
    __shared__ float sh_craw[NN];
    __shared__ float sh_c1[DD];
    __shared__ float sh_c2[DD];
    int b = blockIdx.x, tid = threadIdx.x;

    for (int l = tid; l < LL; l += 256) sh_s[l] = s[b * LL + l];
    __syncthreads();
    // max
    float lm = -3.0e38f;
    for (int l = tid; l < LL; l += 256) lm = fmaxf(lm, sh_s[l]);
    sh_red[tid] = lm; __syncthreads();
    for (int o = 128; o > 0; o >>= 1) { if (tid < o) sh_red[tid] = fmaxf(sh_red[tid], sh_red[tid + o]); __syncthreads(); }
    float mx = sh_red[0]; __syncthreads();
    // exp + sum
    float ls = 0.f;
    for (int l = tid; l < LL; l += 256) { float e = expf(sh_s[l] - mx); sh_s[l] = e; ls += e; }
    sh_red[tid] = ls; __syncthreads();
    for (int o = 128; o > 0; o >>= 1) { if (tid < o) sh_red[tid] += sh_red[tid + o]; __syncthreads(); }
    float inv = 1.f / sh_red[0]; __syncthreads();
    for (int l = tid; l < LL; l += 256) sh_s[l] *= inv;
    __syncthreads();
    // center_raw[n] = sum_l attn[l] * x[b,l,n]
    int n0 = tid, n1 = tid + 256;
    float a0 = 0.f, a1 = 0.f;
    const float* xb = x + (size_t)b * LL * NN;
    for (int l = 0; l < LL; ++l) {
        float a = sh_s[l];
        a0 += a * xb[(size_t)l * NN + n0];
        a1 += a * xb[(size_t)l * NN + n1];
    }
    sh_craw[n0] = a0; sh_craw[n1] = a1;
    __syncthreads();
    if (tid < DD) {
        float acc = bc1[tid];
        for (int j = 0; j < NN; ++j) acc += sh_craw[j] * Wc1[j * DD + tid];
        sh_c1[tid] = gelu_exact(acc);
    }
    __syncthreads();
    if (tid < DD) {
        float acc = bc2[tid];
        #pragma unroll
        for (int j = 0; j < DD; ++j) acc += sh_c1[j] * Wc2[j * DD + tid];
        sh_c2[tid] = gelu_exact(acc);
    }
    __syncthreads();
    {
        float acc0 = bcn[n0], acc1 = bcn[n1];
        #pragma unroll
        for (int j = 0; j < DD; ++j) {
            float c = sh_c2[j];
            acc0 += c * Wcn[j * NN + n0];
            acc1 += c * Wcn[j * NN + n1];
        }
        center[b * NN + n0] = acc0;
        center[b * NN + n1] = acc1;
    }
}

// ---------------- k5: ring gather + GEMM1(gate) + fuse + GEMM2 + residual + layernorm ----------------
__global__ __launch_bounds__(512) void k5_main(
        const float* __restrict__ x, const float* __restrict__ w8, const int* __restrict__ idx8,
        const float* __restrict__ center,
        const unsigned short* __restrict__ wgT_hi, const unsigned short* __restrict__ wgT_lo,
        const unsigned short* __restrict__ wfT_hi, const unsigned short* __restrict__ wfT_lo,
        const float* __restrict__ bg, const float* __restrict__ bfb,
        const float* __restrict__ gamma, const float* __restrict__ beta,
        float* __restrict__ out) {
    __shared__ float xs[16 * 512];          // 32 KB  x rows (also residual)
    __shared__ float ringb[16 * 512];       // 32 KB  ring, later h staging
    __shared__ float fusedb[16 * 512];      // 32 KB  gate-fused rows
    __shared__ unsigned short ahi[16 * 1024];  // 32 KB A hi (bf16, plain k order)
    __shared__ unsigned short alo[16 * 1024];  // 32 KB A lo
    const int tid = threadIdx.x;
    const int row0 = blockIdx.x * 16;

    // load 16 contiguous x rows (8192 contiguous f32)
    const float* xsrc = x + (size_t)row0 * NN;
    for (int i = tid; i < 16 * NN; i += 512) xs[i] = xsrc[i];
    __syncthreads();

    // ring gather + build A1 = [ring | center] as bf16 hi/lo in plain k order (16 x 1024)
    for (int i = tid; i < 16 * 1024; i += 512) {
        int m = i >> 10, k = i & 1023;
        float v;
        if (k < 512) {
            int n = k;
            float acc = 0.f;
            const float* wr = w8 + n * TOPK;
            const int*   ir = idx8 + n * TOPK;
            #pragma unroll
            for (int kk = 0; kk < TOPK; ++kk) acc += wr[kk] * xs[m * 512 + ir[kk]];
            ringb[m * 512 + n] = acc;
            v = acc;
        } else {
            int bm = (row0 + m) / LL;
            v = center[bm * NN + (k - 512)];
        }
        unsigned short hi, lo; split_bf16(v, hi, lo);
        int pos = k;
        ahi[m * 1024 + pos] = hi;
        alo[m * 1024 + pos] = lo;
    }
    __syncthreads();

    const int wave = tid >> 5, lane = tid & 31;
    const int lm = lane & 15, g = lane >> 4;

    // ---- GEMM1: C1 = A1 @ Wg (K=1024), bf16 split: Ah*Bh + Ah*Bl + Al*Bh ----
    for (int t = 0; t < 2; ++t) {
        int nt = wave + 16 * t;
        v8f acc = {0.f, 0.f, 0.f, 0.f, 0.f, 0.f, 0.f, 0.f};
        const unsigned short* arow_h = ahi + lm * 1024;
        const unsigned short* arow_l = alo + lm * 1024;
        const unsigned short* brow_h = wgT_hi + (size_t)(nt * 16 + lm) * 1024;
        const unsigned short* brow_l = wgT_lo + (size_t)(nt * 16 + lm) * 1024;
        for (int ks = 0; ks < 32; ++ks) {
            int off = ks * 32 + g * 16;
            v16bf Ah = load_v16(arow_h + off);
            v16bf Al = load_v16(arow_l + off);
            v16bf Bh = load_v16(brow_h + off);
            v16bf Bl = load_v16(brow_l + off);
            acc = wmma_bf16(Ah, Bh, acc);
            acc = wmma_bf16(Ah, Bl, acc);
            acc = wmma_bf16(Al, Bh, acc);
        }
        int n = nt * 16 + lm;
        float bgn = bg[n];
        #pragma unroll
        for (int r = 0; r < 8; ++r) {
            int m = r + 8 * g;
            float val = acc[r] + bgn;
            float gt = 1.f / (1.f + expf(-val));
            int bm = (row0 + m) / LL;
            float fz = gt * ringb[m * 512 + n] + (1.f - gt) * center[bm * NN + n];
            fusedb[m * 512 + n] = fz;
        }
    }
    __syncthreads();

    // build A2 = fused (16 x 512) bf16 hi/lo in plain k order (reuse ahi/alo, row stride 512)
    for (int i = tid; i < 16 * 512; i += 512) {
        int m = i >> 9, k = i & 511;
        unsigned short hi, lo; split_bf16(fusedb[m * 512 + k], hi, lo);
        int pos = k;
        ahi[m * 512 + pos] = hi;
        alo[m * 512 + pos] = lo;
    }
    __syncthreads();

    // ---- GEMM2: C2 = fused @ Wf (K=512), + bias + residual -> h in ringb ----
    for (int t = 0; t < 2; ++t) {
        int nt = wave + 16 * t;
        v8f acc = {0.f, 0.f, 0.f, 0.f, 0.f, 0.f, 0.f, 0.f};
        const unsigned short* arow_h = ahi + lm * 512;
        const unsigned short* arow_l = alo + lm * 512;
        const unsigned short* brow_h = wfT_hi + (size_t)(nt * 16 + lm) * 512;
        const unsigned short* brow_l = wfT_lo + (size_t)(nt * 16 + lm) * 512;
        for (int ks = 0; ks < 16; ++ks) {
            int off = ks * 32 + g * 16;
            v16bf Ah = load_v16(arow_h + off);
            v16bf Al = load_v16(arow_l + off);
            v16bf Bh = load_v16(brow_h + off);
            v16bf Bl = load_v16(brow_l + off);
            acc = wmma_bf16(Ah, Bh, acc);
            acc = wmma_bf16(Ah, Bl, acc);
            acc = wmma_bf16(Al, Bh, acc);
        }
        int n = nt * 16 + lm;
        float bfn = bfb[n];
        #pragma unroll
        for (int r = 0; r < 8; ++r) {
            int m = r + 8 * g;
            ringb[m * 512 + n] = acc[r] + bfn + xs[m * 512 + n];  // h = fused@Wf + bf + x
        }
    }
    __syncthreads();

    // ---- layernorm: one wave per row ----
    {
        int m = wave;
        const float* hr = ringb + m * 512;
        float sm = 0.f, sq = 0.f;
        #pragma unroll
        for (int j = 0; j < 16; ++j) { float v = hr[j * 32 + lane]; sm += v; sq += v * v; }
        sm = wave_sum(sm); sq = wave_sum(sq);
        float mu = sm * (1.f / 512.f);
        float var = sq * (1.f / 512.f) - mu * mu;
        float rstd = rsqrtf(var + 1e-5f);
        float* orow = out + (size_t)(row0 + m) * NN;
        #pragma unroll
        for (int j = 0; j < 16; ++j) {
            int n = j * 32 + lane;
            orow[n] = (hr[n] - mu) * rstd * gamma[n] + beta[n];
        }
    }
}

// ---------------- launch ----------------
extern "C" void kernel_launch(void* const* d_in, const int* in_sizes, int n_in,
                              void* d_out, int out_size, void* d_ws, size_t ws_size,
                              hipStream_t stream) {
    const float* x      = (const float*)d_in[0];
    const float* ve     = (const float*)d_in[1];
    const float* Wq     = (const float*)d_in[2];
    const float* bq     = (const float*)d_in[3];
    const float* Wk     = (const float*)d_in[4];
    const float* bk     = (const float*)d_in[5];
    const float* Wscore = (const float*)d_in[6];
    const float* bscore = (const float*)d_in[7];
    const float* Wc1    = (const float*)d_in[8];
    const float* bc1    = (const float*)d_in[9];
    const float* Wc2    = (const float*)d_in[10];
    const float* bc2    = (const float*)d_in[11];
    const float* Wcn    = (const float*)d_in[12];
    const float* bcn    = (const float*)d_in[13];
    const float* Wg     = (const float*)d_in[14];
    const float* bg     = (const float*)d_in[15];
    const float* Wf     = (const float*)d_in[16];
    const float* bfb    = (const float*)d_in[17];
    const float* gamma  = (const float*)d_in[18];
    const float* beta   = (const float*)d_in[19];
    float* out = (float*)d_out;

    char* ws = (char*)d_ws;
    size_t off = 0;
    float* Qg   = (float*)(ws + off); off += 512 * 16 * 4;
    float* Kg   = (float*)(ws + off); off += 512 * 16 * 4;
    float* w8   = (float*)(ws + off); off += 512 * 8 * 4;
    int*   idx8 = (int*)  (ws + off); off += 512 * 8 * 4;
    float* s    = (float*)(ws + off); off += ROWS * 4;
    float* cen  = (float*)(ws + off); off += BB * NN * 4;
    unsigned short* wgT_hi = (unsigned short*)(ws + off); off += 512 * 1024 * 2;
    unsigned short* wgT_lo = (unsigned short*)(ws + off); off += 512 * 1024 * 2;
    unsigned short* wfT_hi = (unsigned short*)(ws + off); off += 512 * 512 * 2;
    unsigned short* wfT_lo = (unsigned short*)(ws + off); off += 512 * 512 * 2;

    k0_prep<<<(512 * 1024 + 512 * 512 + 255) / 256, 256, 0, stream>>>(Wg, Wf, wgT_hi, wgT_lo, wfT_hi, wfT_lo);
    k1_qk<<<2, 256, 0, stream>>>(ve, Wq, bq, Wk, bk, Qg, Kg);
    k2_topk<<<2, 256, 0, stream>>>(Qg, Kg, w8, idx8);
    k3_score<<<(ROWS * 32) / 256, 256, 0, stream>>>(x, Wscore, bscore, s);
    k4_center<<<BB, 256, 0, stream>>>(x, s, Wc1, bc1, Wc2, bc2, Wcn, bcn, cen);
    k5_main<<<TILES, 512, 0, stream>>>(x, w8, idx8, cen, wgT_hi, wgT_lo, wfT_hi, wfT_lo,
                                       bg, bfb, gamma, beta, out);
}
